// AttentionBlock_53807350284638
// MI455X (gfx1250) — matrix operand order repaired and checked by reading, not compiled, not run on hardware
//
#include <hip/hip_runtime.h>
#include <hip/hip_bf16.h>
#include <math.h>

typedef _Float16 half_t;
typedef __attribute__((ext_vector_type(16))) _Float16 v16h;
typedef __attribute__((ext_vector_type(8)))  float    v8f;

#define B_  32
#define S_  512
#define H_  256
#define NH_ 8
#define HD_ 32
#define M_  (B_*S_)   /* 16384 rows */

static __device__ __forceinline__ v8f wmma_f16(v16h a, v16h b, v8f c) {
  // v_wmma_f32_16x16x32_f16: D = A(16x32) * B(32x16) + C, fp32 accumulate
  return __builtin_amdgcn_wmma_f32_16x16x32_f16(false, a, false, b, (short)0, c, false, false);
}

// LDS byte offset of a generic pointer: flat->LDS translation truncates to addr[31:0] (ISA 10.2)
static __device__ __forceinline__ unsigned lds_off(const void* p) {
  return (unsigned)(unsigned long long)p;
}

// Async global->LDS 128-bit per-lane copy (GLOBAL_LOAD_ASYNC_TO_LDS_B128, ASYNCcnt-tracked)
static __device__ __forceinline__ void async_copy_b128(unsigned lds_byte_off, const void* gaddr) {
  asm volatile("global_load_async_to_lds_b128 %0, %1, off"
               :: "v"(lds_byte_off), "v"(gaddr) : "memory");
}
static __device__ __forceinline__ void wait_asynccnt0() {
  asm volatile("s_wait_asynccnt 0x0" ::: "memory");
}

// ---- fragment loaders (ISA 7.12.2 layouts, wave32) ----

// A matrix 16x32 f16, row-major source with row stride ld.
// lanes 0-15: M=lane, K-halves {0..7,16..23}; lanes 16-31: M=lane-16, +8.
static __device__ __forceinline__ v16h frag_a_rows(const half_t* p, int ld) {
  int lane = threadIdx.x & 31;
  int m   = lane & 15;
  int khi = (lane >> 4) ? 8 : 0;
  v16h f;
#pragma unroll
  for (int e = 0; e < 16; ++e) {
    int k = ((e < 8) ? e : (e + 8)) + khi;
    f[e] = p[m * ld + k];
  }
  return f;
}

// B matrix 32x16 where column n is ROW n of a row-major [N][K] source.
// lanes 0-15: K-halves {0..7,16..23}; lanes 16-31: +8; N = lane%16.
static __device__ __forceinline__ v16h frag_b_rows(const half_t* p, int ld) {
  int lane = threadIdx.x & 31;
  int n   = lane & 15;
  int khi = (lane >> 4) ? 8 : 0;
  v16h f;
#pragma unroll
  for (int e = 0; e < 16; ++e)
    f[e] = p[n * ld + ((e < 8) ? e : (e + 8)) + khi];
  return f;
}

// ---------------- kernel 0: weight conversion + log-mask ----------------
__global__ void prep_kernel(const float* __restrict__ wqkv, const float* __restrict__ wout,
                            const float* __restrict__ rmask,
                            half_t* __restrict__ wqkv_h, half_t* __restrict__ wout_h,
                            float* __restrict__ logmask) {
  int stride = gridDim.x * blockDim.x;
  int tid = blockIdx.x * blockDim.x + threadIdx.x;
  for (int i = tid; i < 3 * H_ * H_; i += stride) wqkv_h[i] = (half_t)wqkv[i];
  for (int i = tid; i < H_ * H_;     i += stride) wout_h[i] = (half_t)wout[i];
  for (int i = tid; i < M_;          i += stride) logmask[i] = __logf(rmask[i]);
}

// ---------------- kernel 1: LayerNorm -> f16 rows ----------------
__global__ void ln_kernel(const float* __restrict__ x, const float* __restrict__ gamma,
                          const float* __restrict__ beta, half_t* __restrict__ outp) {
  int row = blockIdx.x;
  int t = threadIdx.x;                       // 256 threads, one per element
  float v = x[(size_t)row * H_ + t];
  float s1 = v, s2 = v * v;
#pragma unroll
  for (int off = 16; off > 0; off >>= 1) {
    s1 += __shfl_xor(s1, off);
    s2 += __shfl_xor(s2, off);
  }
  __shared__ float a1[8], a2[8];
  int w = t >> 5;
  if ((t & 31) == 0) { a1[w] = s1; a2[w] = s2; }
  __syncthreads();
  float t1 = 0.f, t2 = 0.f;
#pragma unroll
  for (int i = 0; i < 8; ++i) { t1 += a1[i]; t2 += a2[i]; }
  float mu   = t1 * (1.0f / H_);
  float var  = t2 * (1.0f / H_) - mu * mu;
  float rstd = rsqrtf(var + 1e-5f);
  outp[(size_t)row * H_ + t] = (half_t)((v - mu) * rstd * gamma[t] + beta[t]);
}

// ---------------- kernel 2: QKV GEMM (WMMA) + per-tensor transforms ----------------
// qkv[s][o] = sum_h normed[s][h] * Wqkv[o][h];  M=16384, N=768, K=256
__global__ void qkv_kernel(const half_t* __restrict__ normed, const half_t* __restrict__ wqkv_h,
                           const float* __restrict__ logmask,
                           half_t* __restrict__ qh, half_t* __restrict__ kh, half_t* __restrict__ vh) {
  __shared__ __align__(16) half_t As[64 * 32];
  __shared__ __align__(16) half_t Bs[64 * 32];
  int m0 = blockIdx.x * 64;
  int n0 = blockIdx.y * 64;
  int t = threadIdx.x;
  int w = t >> 5, lane = t & 31;
  int mt = w & 3;               // M sub-tile 0..3
  int nt0 = (w >> 2) * 2;       // N sub-tiles {0,1} or {2,3}
  v8f c0 = {}; v8f c1 = {};
  for (int k0 = 0; k0 < H_; k0 += 32) {
    __syncthreads();
    int r = t >> 2, cc = (t & 3) * 8;   // 16B per thread
    *(uint4*)(&As[r * 32 + cc]) = *(const uint4*)(&normed[(size_t)(m0 + r) * H_ + k0 + cc]);
    *(uint4*)(&Bs[r * 32 + cc]) = *(const uint4*)(&wqkv_h[(size_t)(n0 + r) * H_ + k0 + cc]);
    __syncthreads();
    v16h a  = frag_a_rows(&As[mt * 16 * 32], 32);
    v16h b0 = frag_b_rows(&Bs[nt0 * 16 * 32], 32);
    v16h b1 = frag_b_rows(&Bs[(nt0 + 1) * 16 * 32], 32);
    c0 = wmma_f16(a, b0, c0);
    c1 = wmma_f16(a, b1, c1);
  }
  int n = lane & 15;
  int hi8 = (lane >> 4) * 8;
#pragma unroll
  for (int sub = 0; sub < 2; ++sub) {
    v8f c = sub ? c1 : c0;
    int o = n0 + (nt0 + sub) * 16 + n;
#pragma unroll
    for (int r = 0; r < 8; ++r) {
      int row = m0 + mt * 16 + r + hi8;
      float val = c[r];
      half_t* dst; int oo;
      if (o < H_)            { val = val * val + 1e-6f; dst = qh; oo = o; }          // q = q*q+eps
      else if (o < 2 * H_)   { val += logmask[row];     dst = kh; oo = o - H_; }     // k += log(mask)
      else                   {                          dst = vh; oo = o - 2 * H_; } // v
      int head = oo >> 5, d = oo & 31;
      int bb = row >> 9, s = row & 511;
      dst[(((size_t)(bb * NH_ + head) * S_) + s) * HD_ + d] = (half_t)val;
    }
  }
}

// ---------------- kernel 3: attention core (per head, 64-query tile) ----------------
__global__ void attn_kernel(const half_t* __restrict__ qh, const half_t* __restrict__ kh,
                            const half_t* __restrict__ vh, half_t* __restrict__ attnout) {
  extern __shared__ char smem[];
  half_t* qs = (half_t*)smem;                 // 64 x 32 (row-major)
  half_t* ks = qs + 64 * 32;                  // 512 x 32 (row-major)
  half_t* vt = ks + 512 * 32;                 // 32 x 512 (V TRANSPOSED: vt[d][k])
  float*  sc = (float*)(vt + 32 * 512);       // 64 x 512 fp32 scores
  half_t* ps = (half_t*)(sc + 64 * 512);      // 64 x 512 f16 probs (unnormalized)
  float*  rs = (float*)(ps + 64 * 512);       // 64 row sums
  int bh = blockIdx.x;                        // b*NH + h
  int q0 = blockIdx.y * 64;
  int t = threadIdx.x;
  const half_t* qbase = qh + (size_t)bh * S_ * HD_;
  const half_t* kbase = kh + (size_t)bh * S_ * HD_;
  const half_t* vbase = vh + (size_t)bh * S_ * HD_;

  // Q tile (4 KB) and K (32 KB): async global->LDS, 128b per lane, ASYNCcnt-tracked.
  async_copy_b128(lds_off(&qs[t * 8]), &qbase[(size_t)q0 * HD_ + t * 8]);
#pragma unroll
  for (int i = 0; i < 8; ++i) {
    int idx = (t + i * 256) * 8;
    async_copy_b128(lds_off(&ks[idx]), &kbase[idx]);
  }
  // V: load row, store transposed into vt[d][k] so the P*V B-fragment is row-contiguous.
#pragma unroll
  for (int rr = 0; rr < 2; ++rr) {
    int r = t + rr * 256;                     // key index 0..511
    const uint4* src = (const uint4*)&vbase[(size_t)r * HD_];
    uint4 u[4] = { src[0], src[1], src[2], src[3] };
    const half_t* hrow = (const half_t*)u;
#pragma unroll
    for (int d = 0; d < 32; ++d) vt[d * 512 + r] = hrow[d];
  }
  wait_asynccnt0();
  __syncthreads();

  int w = t >> 5, lane = t & 31;
  int n = lane & 15, hi8 = (lane >> 4) * 8;
  const float scale = 0.17677669529663687f;   // 1/sqrt(32)
  // scores = q @ k^T : 4 x 32 tiles of 16x16, one K-step (hd=32) each; 16 tiles/wave
#pragma unroll 4
  for (int i = 0; i < 16; ++i) {
    int tile = w * 16 + i;
    int mt = tile >> 5, nt = tile & 31;
    v16h a = frag_a_rows(&qs[mt * 16 * 32], 32);
    v16h b = frag_b_rows(&ks[nt * 16 * 32], 32);
    v8f c = {};
    c = wmma_f16(a, b, c);
#pragma unroll
    for (int r = 0; r < 8; ++r)
      sc[(mt * 16 + r + hi8) * 512 + nt * 16 + n] = c[r] * scale;
  }
  __syncthreads();
  // softmax: 4 threads per row (consecutive lanes), shuffle-combine
  {
    int row = t >> 2, sub = t & 3;
    float m = -1e30f;
    for (int j = sub; j < 512; j += 4) m = fmaxf(m, sc[row * 512 + j]);
    m = fmaxf(m, __shfl_xor(m, 1));
    m = fmaxf(m, __shfl_xor(m, 2));
    float sum = 0.f;
    for (int j = sub; j < 512; j += 4) {
      float e = __expf(sc[row * 512 + j] - m);
      ps[row * 512 + j] = (half_t)e;
      sum += e;
    }
    sum += __shfl_xor(sum, 1);
    sum += __shfl_xor(sum, 2);
    if (sub == 0) rs[row] = sum;
  }
  __syncthreads();
  // out = P @ V : M=64, N=32, K=512 -> 8 tiles, one per wave; 32 K-steps.
  // B frag: B[k][n] = v[k][d] = vt[d][k] -> row-contiguous in vt.
  {
    int mt = w >> 1, nt = w & 1;
    v8f c = {};
    for (int kk = 0; kk < 32; ++kk) {
      v16h a = frag_a_rows(&ps[(size_t)mt * 16 * 512 + kk * 32], 512);
      v16h b = frag_b_rows(&vt[(size_t)(nt * 16) * 512 + kk * 32], 512);
      c = wmma_f16(a, b, c);
    }
    int bb = bh >> 3, h = bh & 7;
#pragma unroll
    for (int r = 0; r < 8; ++r) {
      int mrow = mt * 16 + r + hi8;
      int s = q0 + mrow;
      float inv = 1.0f / rs[mrow];
      attnout[((size_t)(bb * S_ + s)) * H_ + h * HD_ + nt * 16 + n] = (half_t)(c[r] * inv);
    }
  }
}

// ---------------- kernel 4: out projection + residual ----------------
// y = attnout @ Wout^T ; out = (y + inputs) * sqrt(0.5)
__global__ void proj_kernel(const half_t* __restrict__ attnout, const half_t* __restrict__ wout_h,
                            const float* __restrict__ inputs, float* __restrict__ out) {
  __shared__ __align__(16) half_t As[64 * 32];
  __shared__ __align__(16) half_t Bs[64 * 32];
  int m0 = blockIdx.x * 64;
  int n0 = blockIdx.y * 64;
  int t = threadIdx.x;
  int w = t >> 5, lane = t & 31;
  int mt = w & 3, nt0 = (w >> 2) * 2;
  v8f c0 = {}, c1 = {};
  for (int k0 = 0; k0 < H_; k0 += 32) {
    __syncthreads();
    int r = t >> 2, cc = (t & 3) * 8;
    *(uint4*)(&As[r * 32 + cc]) = *(const uint4*)(&attnout[(size_t)(m0 + r) * H_ + k0 + cc]);
    *(uint4*)(&Bs[r * 32 + cc]) = *(const uint4*)(&wout_h[(size_t)(n0 + r) * H_ + k0 + cc]);
    __syncthreads();
    v16h a = frag_a_rows(&As[mt * 16 * 32], 32);
    c0 = wmma_f16(a, frag_b_rows(&Bs[nt0 * 16 * 32], 32), c0);
    c1 = wmma_f16(a, frag_b_rows(&Bs[(nt0 + 1) * 16 * 32], 32), c1);
  }
  int n = lane & 15, hi8 = (lane >> 4) * 8;
  const float rsqrt2 = 0.70710678118654752f;
#pragma unroll
  for (int sub = 0; sub < 2; ++sub) {
    v8f c = sub ? c1 : c0;
    int o = n0 + (nt0 + sub) * 16 + n;
#pragma unroll
    for (int r = 0; r < 8; ++r) {
      int row = m0 + mt * 16 + r + hi8;
      out[(size_t)row * H_ + o] = (c[r] + inputs[(size_t)row * H_ + o]) * rsqrt2;
    }
  }
}

extern "C" void kernel_launch(void* const* d_in, const int* in_sizes, int n_in,
                              void* d_out, int out_size, void* d_ws, size_t ws_size,
                              hipStream_t stream) {
  const float* inputs = (const float*)d_in[0];
  const float* rmask  = (const float*)d_in[1];
  const float* wqkv   = (const float*)d_in[2];
  const float* wout   = (const float*)d_in[3];
  const float* gamma  = (const float*)d_in[4];
  const float* beta   = (const float*)d_in[5];
  float* out = (float*)d_out;

  char* ws = (char*)d_ws;
  size_t off = 0;
  half_t* wqkv_h  = (half_t*)(ws + off); off += (size_t)3 * H_ * H_ * sizeof(half_t);
  half_t* wout_h  = (half_t*)(ws + off); off += (size_t)H_ * H_ * sizeof(half_t);
  float*  logmask = (float*) (ws + off); off += (size_t)M_ * sizeof(float);
  half_t* normed  = (half_t*)(ws + off); off += (size_t)M_ * H_ * sizeof(half_t);
  half_t* attnout = normed;  // normed is dead after qkv_kernel -> reuse
  half_t* qh = (half_t*)(ws + off); off += (size_t)M_ * H_ * sizeof(half_t);
  half_t* kh = (half_t*)(ws + off); off += (size_t)M_ * H_ * sizeof(half_t);
  half_t* vh = (half_t*)(ws + off); off += (size_t)M_ * H_ * sizeof(half_t);

  prep_kernel<<<dim3(256), dim3(256), 0, stream>>>(wqkv, wout, rmask, wqkv_h, wout_h, logmask);
  ln_kernel<<<dim3(M_), dim3(256), 0, stream>>>(inputs, gamma, beta, normed);
  qkv_kernel<<<dim3(M_ / 64, (3 * H_) / 64), dim3(256), 0, stream>>>(normed, wqkv_h, logmask,
                                                                     qh, kh, vh);
  size_t smem = (size_t)64 * 32 * 2 + (size_t)512 * 32 * 2 * 2 +
                (size_t)64 * 512 * 4 + (size_t)64 * 512 * 2 + 64 * 4; // 266,496 B (< 320 KB WGP LDS)
  hipFuncSetAttribute((const void*)attn_kernel,
                      hipFuncAttributeMaxDynamicSharedMemorySize, (int)smem);
  attn_kernel<<<dim3(B_ * NH_, S_ / 64), dim3(256), smem, stream>>>(qh, kh, vh, attnout);
  proj_kernel<<<dim3(M_ / 64, H_ / 64), dim3(256), 0, stream>>>(attnout, wout_h, inputs, out);
}
